// MaxSimInfoNCE_79078937854110
// MI455X (gfx1250) — matrix operand order repaired and checked by reading, not compiled, not run on hardware
//
#include <hip/hip_runtime.h>
#include <hip/hip_bf16.h>
#include <stdint.h>

// ---------------------------------------------------------------------------
// MaxSim InfoNCE for MI455X (gfx1250, wave32, WMMA)
//   sim[b1,b2] = mean_t max_i <text[b1,t,:], image[b2,i,:]>
//   loss = 0.5*(CE(sim/T) + CE(sim^T/T)) with diagonal targets
// B=128, T=64, I=196 (pad->208), D=512
// ---------------------------------------------------------------------------

typedef __attribute__((ext_vector_type(16))) __bf16 v16bf;
typedef __attribute__((ext_vector_type(8)))  __bf16 v8bf;
typedef __attribute__((ext_vector_type(4)))  __bf16 v4bf;
typedef __attribute__((ext_vector_type(8)))  float  v8f;

#define NB   128
#define NT   64
#define NI   196
#define NIP  208     // padded image tokens (13 tiles of 16)
#define ND   512
#define KC   32      // K chunk held in LDS per stage
#define NKC  (ND / KC)
#define NEGF (-3.0e38f)

// ---------------- CDNA5 async global->LDS copy (ASYNCcnt) -------------------
__device__ __forceinline__ void async_copy_b128(const void* gptr, unsigned lds_off) {
  unsigned long long ga = (unsigned long long)gptr;
  asm volatile("global_load_async_to_lds_b128 %0, %1, off"
               :: "v"(lds_off), "v"(ga) : "memory");
}
__device__ __forceinline__ void wait_async0() {
  asm volatile("s_wait_asynccnt 0" ::: "memory");
}
__device__ __forceinline__ unsigned lds_offset(const void* p) {
  // generic pointer to __shared__ = {SHARED_BASE aperture, LDS byte offset}
  return (unsigned)(unsigned long long)p;
}

// ---------------- fp32 -> bf16 conversion kernels ---------------------------
__global__ __launch_bounds__(256) void cvt_text_kernel(const float* __restrict__ in,
                                                       __bf16* __restrict__ out, int n4) {
  int i = blockIdx.x * blockDim.x + threadIdx.x;
  if (i >= n4) return;
  float4 v = ((const float4*)in)[i];
  v4bf o = { (__bf16)v.x, (__bf16)v.y, (__bf16)v.z, (__bf16)v.w };
  ((v4bf*)out)[i] = o;
}

__global__ __launch_bounds__(256) void cvt_image_kernel(const float* __restrict__ in,
                                                        __bf16* __restrict__ out) {
  int i  = blockIdx.x * blockDim.x + threadIdx.x;   // over 128*208*(512/4)
  int d4 = i & 127;                                  // 512/4 = 128
  int r  = (i >> 7) % NIP;
  int b  = i / (NIP * 128);
  if (b >= NB) return;
  v4bf o = { (__bf16)0.f, (__bf16)0.f, (__bf16)0.f, (__bf16)0.f };
  if (r < NI) {
    float4 v = ((const float4*)in)[(size_t)(b * NI + r) * 128 + d4];
    o.x = (__bf16)v.x; o.y = (__bf16)v.y; o.z = (__bf16)v.z; o.w = (__bf16)v.w;
  }
  ((v4bf*)out)[(size_t)(b * NIP + r) * 128 + d4] = o;
}

// ---------------- stage one K-chunk of A (text) and B (image) into LDS ------
__device__ __forceinline__ void stage_chunk(const __bf16* __restrict__ txtb,
                                            const __bf16* __restrict__ imgb,
                                            int b1, int b2, int kc,
                                            __bf16* Ab, __bf16* Bb, int tid) {
  // A: 64 rows x 32 cols  -> 256 chunks of 8 bf16 (16B)
  // B: 208 rows x 32 cols -> 832 chunks
  for (int i = tid; i < 256 + 832; i += 256) {
    if (i < 256) {
      int row = i >> 2, c = i & 3;
      const __bf16* g = txtb + (size_t)(b1 * NT + row) * ND + kc + c * 8;
      async_copy_b128(g, lds_offset(Ab + row * KC + c * 8));
    } else {
      int j = i - 256;
      int row = j >> 2, c = j & 3;
      const __bf16* g = imgb + (size_t)(b2 * NIP + row) * ND + kc + c * 8;
      async_copy_b128(g, lds_offset(Bb + row * KC + c * 8));
    }
  }
}

// ---------------- main MaxSim kernel: one block per (b1,b2) pair ------------
__global__ __launch_bounds__(256) void maxsim_kernel(const __bf16* __restrict__ txtb,
                                                     const __bf16* __restrict__ imgb,
                                                     float* __restrict__ sim) {
  __shared__ __bf16 Ab[2][NT  * KC];   // 2 x 4KB
  __shared__ __bf16 Bb[2][NIP * KC];   // 2 x 13KB
  __shared__ float  partialmax[NT][2];
  __shared__ float  rowred[NT];

  const int b2 = blockIdx.x;           // image batch
  const int b1 = blockIdx.y;           // text batch
  const int tid  = threadIdx.x;
  const int lane = tid & 31;
  const int w    = tid >> 5;           // wave 0..7
  const int mt   = w >> 1;             // M tile 0..3 (text rows 16*mt..)
  const int ncol = w & 1;              // N-tile half
  const int nt0  = ncol ? 7 : 0;       // this wave's first N tile (of 13)

  v8f acc[7] = {};                     // up to 7 16x16 f32 accumulators

  stage_chunk(txtb, imgb, b1, b2, 0, Ab[0], Bb[0], tid);
  wait_async0();
  __syncthreads();

  for (int kci = 0; kci < NKC; ++kci) {
    const int cur = kci & 1;
    if (kci + 1 < NKC)
      stage_chunk(txtb, imgb, b1, b2, (kci + 1) * KC, Ab[cur ^ 1], Bb[cur ^ 1], tid);

    const __bf16* As = Ab[cur];
    const __bf16* Bs = Bb[cur];

    // A fragment (16x32 bf16): lanes 0-15 hold K{0..15}, lanes 16-31 K{16..31}
    const int arow = mt * 16 + (lane & 15);
    const int ak   = (lane >> 4) << 4;
    v8bf alo = *(const v8bf*)(As + arow * KC + ak);
    v8bf ahi = *(const v8bf*)(As + arow * KC + ak + 8);
    v16bf a = __builtin_shufflevector(alo, ahi, 0,1,2,3,4,5,6,7,8,9,10,11,12,13,14,15);

    // Preload ALL B fragments for this K-step first so the ds_loads batch up
    // and drain with partial dscnt waits instead of serializing per-WMMA.
    v16bf bfrag[7];
    #pragma unroll
    for (int t = 0; t < 7; ++t) {
      const int nt = nt0 + t;
      if (nt < 13) {
        // B fragment (32x16 bf16): lane holds column nt*16+(lane&15),
        // 16 consecutive K starting at (lane>=16 ? 16 : 0)
        const int bn = nt * 16 + (lane & 15);
        const int bk = (lane >> 4) << 4;
        const __bf16* bp = Bs + bn * KC + bk;
        v8bf blo = *(const v8bf*)bp;
        v8bf bhi = *(const v8bf*)(bp + 8);
        bfrag[t] = __builtin_shufflevector(blo, bhi, 0,1,2,3,4,5,6,7,8,9,10,11,12,13,14,15);
      }
    }
    #pragma unroll
    for (int t = 0; t < 7; ++t) {
      const int nt = nt0 + t;
      if (nt < 13) {
        acc[t] = __builtin_amdgcn_wmma_f32_16x16x32_bf16(
            false, a, false, bfrag[t], (short)0, acc[t], false, false);
      }
    }
    wait_async0();
    __syncthreads();
  }

  // ---- epilogue: masked max over image tokens, then mean over text tokens --
  float pm[8];
  #pragma unroll
  for (int r = 0; r < 8; ++r) pm[r] = NEGF;

  #pragma unroll
  for (int t = 0; t < 7; ++t) {
    const int nt = nt0 + t;
    if (nt < 13) {
      const int gcol = nt * 16 + (lane & 15);
      const bool valid = gcol < NI;          // mask zero-padded image tokens
      #pragma unroll
      for (int r = 0; r < 8; ++r) {
        float v = valid ? acc[t][r] : NEGF;
        pm[r] = fmaxf(pm[r], v);
      }
    }
  }
  // reduce max across the 16 lanes of each half-wave (rows r / r+8)
  #pragma unroll
  for (int r = 0; r < 8; ++r) {
    #pragma unroll
    for (int s = 1; s < 16; s <<= 1)
      pm[r] = fmaxf(pm[r], __shfl_xor(pm[r], s, 32));
  }
  if ((lane & 15) == 0) {
    const int rowbase = mt * 16 + ((lane >> 4) << 3);
    #pragma unroll
    for (int r = 0; r < 8; ++r) partialmax[rowbase + r][ncol] = pm[r];
  }
  __syncthreads();
  if (tid < NT) rowred[tid] = fmaxf(partialmax[tid][0], partialmax[tid][1]);
  __syncthreads();
  if (tid == 0) {
    float s = 0.f;
    for (int r = 0; r < NT; ++r) s += rowred[r];
    sim[b1 * NB + b2] = s * (1.0f / NT);
  }
}

// ---------------- scalar InfoNCE loss over the 128x128 sim matrix ----------
__global__ __launch_bounds__(128) void loss_kernel(const float* __restrict__ sim,
                                                   float* __restrict__ out) {
  __shared__ float red[NB];
  const int b = threadIdx.x;
  const float invT = 1.0f / 0.07f;
  float mr = NEGF, mc = NEGF;
  for (int j = 0; j < NB; ++j) {
    mr = fmaxf(mr, sim[b * NB + j]);
    mc = fmaxf(mc, sim[j * NB + b]);
  }
  float sr = 0.f, sc = 0.f;
  for (int j = 0; j < NB; ++j) {
    sr += __expf((sim[b * NB + j] - mr) * invT);
    sc += __expf((sim[j * NB + b] - mc) * invT);
  }
  const float diag = sim[b * NB + b] * invT;
  const float lr = mr * invT + __logf(sr);
  const float lc = mc * invT + __logf(sc);
  red[b] = (lr - diag) + (lc - diag);
  __syncthreads();
  if (b == 0) {
    float s = 0.f;
    for (int j = 0; j < NB; ++j) s += red[j];
    out[0] = s * (0.5f / (float)NB);
  }
}

// ---------------------------------------------------------------------------
extern "C" void kernel_launch(void* const* d_in, const int* in_sizes, int n_in,
                              void* d_out, int out_size, void* d_ws, size_t ws_size,
                              hipStream_t stream) {
  (void)in_sizes; (void)n_in; (void)out_size; (void)ws_size;
  const float* img = (const float*)d_in[0];   // [128,196,512] fp32
  const float* txt = (const float*)d_in[1];   // [128,64,512]  fp32
  float* out = (float*)d_out;                 // scalar fp32

  char* ws = (char*)d_ws;
  const size_t txtb_bytes = (size_t)NB * NT  * ND * 2;  //  8,388,608
  const size_t imgb_bytes = (size_t)NB * NIP * ND * 2;  // 27,262,976
  __bf16* txtb = (__bf16*)ws;
  __bf16* imgb = (__bf16*)(ws + txtb_bytes);
  float*  simm = (float*)(ws + txtb_bytes + imgb_bytes);

  // 1) fp32 -> bf16 (image padded 196 -> 208 with zeros)
  {
    int n4 = NB * NT * ND / 4;                      // 1,048,576
    cvt_text_kernel<<<n4 / 256, 256, 0, stream>>>(txt, txtb, n4);
    int n4i = NB * NIP * (ND / 4);                  // 3,407,872
    cvt_image_kernel<<<n4i / 256, 256, 0, stream>>>(img, imgb);
  }

  // 2) per-pair MaxSim GEMMs with bf16 WMMA + async LDS double buffering
  {
    dim3 grid(NB, NB);
    maxsim_kernel<<<grid, 256, 0, stream>>>(txtb, imgb, simm);
  }

  // 3) symmetric InfoNCE loss
  loss_kernel<<<1, 128, 0, stream>>>(simm, out);
}
